// Conv2d_12446815224053
// MI455X (gfx1250) — matrix operand order repaired and checked by reading, not compiled, not run on hardware
//
#include <hip/hip_runtime.h>

// CDNA5 / gfx1250 implicit-GEMM 3x3 conv using split-precision bf16x3 WMMA:
//   f32 operand v = hi + lo  (hi = bf16(v), lo = bf16(v - hi))
//   A*B ~= Ah*Bh + Ah*Bl + Al*Bh   (f32 accumulation, V_WMMA_F32_16X16X32_BF16)
// ~2.7x fewer matrix instructions than the exact f32 16x16x4 path, ~1e-4 rel err.
//
// x: (32,128,56,56) f32 NCHW; weight: (256,128,3,3) f32 OIHW; bias: (256,) f32.
// out: (32,256,56,56) f32.
//
// Pre-passes (into d_ws):
//   1) x -> zero-padded NHWC bf16 hi/lo:  xh/xl[n][58][58][128]
//      (K=32 channel fragments become contiguous 32B loads; no halo masking)
//   2) weight -> split bf16 hi/lo packed in exact A-fragment VGPR order.
// Workspace: 2*13,778,944*2 + 2*294,912*2 = 56,295,424 bytes.
//
// Main kernel: one wave computes a 32(o) x 64(pixel) tile (8 f32 accumulators);
// per K=32 chunk: 8 A b128-loads + 16 B b128-loads feed 24 WMMAs (1.0 load/WMMA).
// 8 waves/block cover all 256 output channels of one 64-pixel tile.

typedef __bf16 v16bf __attribute__((ext_vector_type(16)));
typedef float  v8f   __attribute__((ext_vector_type(8)));

constexpr int Hc = 56, Wc = 56, HWc = 56 * 56;
constexpr int Hp = 58, Wp = 58;                 // zero-padded
constexpr int CI = 128, CO = 256;
constexpr size_t XPAD_ELEMS = (size_t)32 * Hp * Wp * CI;    // 13,778,944
constexpr size_t WPK_ELEMS  = (size_t)16 * 9 * 4 * 32 * 16; // 294,912
constexpr int WT_MT = 9 * 4 * 32 * 16;                      // 18432 elems per m_tile

// ---------------------------------------------------------------------------
// x (NCHW f32) -> padded NHWC bf16 hi/lo. Border written as zero.
// ---------------------------------------------------------------------------
__global__ __launch_bounds__(256) void split_pad_x(
    const float* __restrict__ x, __bf16* __restrict__ xh, __bf16* __restrict__ xl) {
  size_t idx = (size_t)blockIdx.x * 256 + threadIdx.x;
  if (idx >= XPAD_ELEMS) return;
  int c = (int)(idx & (CI - 1));
  int r = (int)(idx >> 7);
  int wp = r % Wp; r /= Wp;
  int hp = r % Hp;
  int n  = r / Hp;
  float v = 0.0f;
  if (hp >= 1 && hp <= Hc && wp >= 1 && wp <= Wc)
    v = x[(((size_t)n * CI + c) * Hc + (hp - 1)) * Wc + (wp - 1)];
  __bf16 h = (__bf16)v;
  __bf16 l = (__bf16)(v - (float)h);
  xh[idx] = h;
  xl[idx] = l;
}

// ---------------------------------------------------------------------------
// Weight (OIHW f32) -> split bf16 hi/lo, packed fragment-linear:
// layout [m_tile(16)][tap(9)][chunk(4)][lane(32)][slot(16)], slot = v*2+h maps
// to A-fragment VGPR v, 16-bit half h. K order matches B (contiguous):
//   k_local = (lane>=16 ? 16:0) + slot,  M = lane&15.
// ---------------------------------------------------------------------------
__global__ __launch_bounds__(256) void split_pack_w(
    const float* __restrict__ wgt, __bf16* __restrict__ wh, __bf16* __restrict__ wl) {
  int idx = blockIdx.x * 256 + threadIdx.x;
  if (idx >= (int)WPK_ELEMS) return;
  int slot  = idx & 15;
  int lane  = (idx >> 4) & 31;
  int chunk = (idx >> 9) & 3;
  int r     = idx >> 11;        // = m_tile*9 + tap
  int tap   = r % 9;
  int mt    = r / 9;
  int v = slot >> 1, hb = slot & 1;
  int kloc = ((lane >= 16) ? 16 : 0) + v * 2 + hb;
  int o = mt * 16 + (lane & 15);
  int c = chunk * 32 + kloc;
  int kh = tap / 3, kw = tap - kh * 3;
  float wv = wgt[((o * CI + c) * 3 + kh) * 3 + kw];
  __bf16 hi = (__bf16)wv;
  __bf16 lo = (__bf16)(wv - (float)hi);
  wh[idx] = hi;
  wl[idx] = lo;
}

// ---------------------------------------------------------------------------
// Main conv. One wave -> 32(o) x 64(pixel) tile: two 16-row o-subtiles
// (acc0/acc1) sharing every B fragment. Grid = 32*49 = 1568 blocks.
// ---------------------------------------------------------------------------
__global__ __launch_bounds__(256) void conv3x3_wmma_bf16x3(
    const __bf16* __restrict__ xh, const __bf16* __restrict__ xl,
    const __bf16* __restrict__ wh, const __bf16* __restrict__ wl,
    const float* __restrict__ bias, float* __restrict__ out) {
  const int lane = threadIdx.x & 31;
  const int wv_  = threadIdx.x >> 5;          // 0..7 -> o base = wv_*32
  const int mt0  = wv_ * 2;                   // first 16-row o-subtile
  const int pix_tile = blockIdx.x;            // 0..1567
  const int n_img = pix_tile / 49;
  const int p_base = (pix_tile - n_img * 49) << 6;   // 64-pixel tile base

  const int m = lane & 15;        // pixel within sub-tile (B/C/D N index)
  const int khalf = lane >> 4;    // which 16-wide K half this lane holds (B)

  int ph[4], pw[4];
#pragma unroll
  for (int s = 0; s < 4; ++s) {
    int p = p_base + (s << 4) + m;
    ph[s] = p / Wc;
    pw[s] = p - ph[s] * Wc;
  }

  const size_t nbase = (size_t)n_img * Hp * Wp * CI;
  v8f acc0[4] = {};
  v8f acc1[4] = {};

  const __bf16* wah = wh + (size_t)mt0 * WT_MT + lane * 16;  // mt1 at +WT_MT
  const __bf16* wal = wl + (size_t)mt0 * WT_MT + lane * 16;

  for (int tap = 0; tap < 9; ++tap) {
    const int kh = tap / 3, kw = tap - kh * 3;
    const __bf16* pxh[4];
    const __bf16* pxl[4];
#pragma unroll
    for (int s = 0; s < 4; ++s) {
      size_t rb = nbase + ((size_t)(ph[s] + kh) * Wp + (pw[s] + kw)) * CI + khalf * 16;
      pxh[s] = xh + rb;
      pxl[s] = xl + rb;
    }
#pragma unroll
    for (int chunk = 0; chunk < 4; ++chunk) {
      const int wofs = tap * 2048 + chunk * 512;
      v16bf Ah0 = *(const v16bf*)(wah + wofs);
      v16bf Al0 = *(const v16bf*)(wal + wofs);
      v16bf Ah1 = *(const v16bf*)(wah + wofs + WT_MT);
      v16bf Al1 = *(const v16bf*)(wal + wofs + WT_MT);
#pragma unroll
      for (int s = 0; s < 4; ++s) {
        v16bf Bh = *(const v16bf*)(pxh[s] + chunk * 32);
        v16bf Bl = *(const v16bf*)(pxl[s] + chunk * 32);
        acc0[s] = __builtin_amdgcn_wmma_f32_16x16x32_bf16(
            false, Ah0, false, Bh, (short)0, acc0[s], false, false);
        acc0[s] = __builtin_amdgcn_wmma_f32_16x16x32_bf16(
            false, Ah0, false, Bl, (short)0, acc0[s], false, false);
        acc0[s] = __builtin_amdgcn_wmma_f32_16x16x32_bf16(
            false, Al0, false, Bh, (short)0, acc0[s], false, false);
        acc1[s] = __builtin_amdgcn_wmma_f32_16x16x32_bf16(
            false, Ah1, false, Bh, (short)0, acc1[s], false, false);
        acc1[s] = __builtin_amdgcn_wmma_f32_16x16x32_bf16(
            false, Ah1, false, Bl, (short)0, acc1[s], false, false);
        acc1[s] = __builtin_amdgcn_wmma_f32_16x16x32_bf16(
            false, Al1, false, Bh, (short)0, acc1[s], false, false);
      }
    }
  }

  // Writeback with bias. C/D 16x16 f32 layout: VGPR v -> M = v + (lane>>4)*8.
#pragma unroll
  for (int t = 0; t < 2; ++t) {
#pragma unroll
    for (int v = 0; v < 8; ++v) {
      int o = (mt0 + t) * 16 + v + (khalf << 3);
      float bv = bias[o];
      float* po = out + ((size_t)(n_img * CO + o)) * HWc + p_base + m;
#pragma unroll
      for (int s = 0; s < 4; ++s)
        po[s << 4] = (t == 0 ? acc0[s][v] : acc1[s][v]) + bv;
    }
  }
}

// ---------------------------------------------------------------------------
extern "C" void kernel_launch(void* const* d_in, const int* in_sizes, int n_in,
                              void* d_out, int out_size, void* d_ws, size_t ws_size,
                              hipStream_t stream) {
  const float* x    = (const float*)d_in[0];  // 32*128*56*56
  const float* wgt  = (const float*)d_in[1];  // 256*128*3*3
  const float* bias = (const float*)d_in[2];  // 256
  float* out = (float*)d_out;                 // 32*256*56*56

  __bf16* xh = (__bf16*)d_ws;
  __bf16* xl = xh + XPAD_ELEMS;
  __bf16* wh = xl + XPAD_ELEMS;
  __bf16* wl = wh + WPK_ELEMS;                // total ws use: 56,295,424 bytes

  (void)in_sizes; (void)n_in; (void)out_size; (void)ws_size;

  split_pad_x<<<(int)((XPAD_ELEMS + 255) / 256), 256, 0, stream>>>(x, xh, xl);
  split_pack_w<<<(int)((WPK_ELEMS + 255) / 256), 256, 0, stream>>>(wgt, wh, wl);
  conv3x3_wmma_bf16x3<<<32 * 49, 256, 0, stream>>>(xh, xl, wh, wl, bias, out);
}
